// WanRefAttnProcessor_44976897524323
// MI455X (gfx1250) — matrix operand order repaired and checked by reading, not compiled, not run on hardware
//
#include <hip/hip_runtime.h>

// ---------------------------------------------------------------------------
// Types for CDNA5 WMMA (wave32): 16x16x32 bf16 -> f32
// ---------------------------------------------------------------------------
typedef __attribute__((ext_vector_type(16))) __bf16 v16bf;
typedef __attribute__((ext_vector_type(8)))  __bf16 v8bf;
typedef __attribute__((ext_vector_type(8)))  float  v8f;
typedef __attribute__((ext_vector_type(4)))  unsigned int v4u;
typedef __attribute__((ext_vector_type(8)))  unsigned int v8u;

#define DEVFN __device__ __forceinline__

namespace {
constexpr int kT   = 4096;   // main sequence length
constexpr int kTR  = 1024;   // reference sequence length
constexpr int kDIM = 1536;
constexpr int kH   = 12;
constexpr int kDH  = 128;    // head dim
}

// ---------------------------------------------------------------------------
// Fragment loaders (ISA 7.12.2 layouts, wave32)
// A (16x32, 16-bit): lane L: row M=L&15, elems 0..7 -> K=kb+8*hf+e,
// elems 8..15 -> K=kb+16+8*hf+(e-8), hf=L>>4.  Two 16B chunks per lane.
// ---------------------------------------------------------------------------
DEVFN v16bf load_a16x32(const __bf16* base, int ld, int kb) {
  const int lane = threadIdx.x & 31;
  const int hf = lane >> 4;
  const int r  = lane & 15;
  const __bf16* p = base + (size_t)r * ld + kb + hf * 8;
  v8bf lo = *reinterpret_cast<const v8bf*>(p);
  v8bf hi = *reinterpret_cast<const v8bf*>(p + 16);
  v16bf out;
#pragma unroll
  for (int i = 0; i < 8; ++i) { out[i] = lo[i]; out[8 + i] = hi[i]; }
  return out;
}

// B (32x16, 16-bit) from a K-contiguous matrix Bt[N][K]:
// lane L: col N = n0 + (L&15), elems 0..7 -> K = kb + 8*hf + e,
// elems 8..15 -> K = kb + 16 + 8*hf + (e-8). Two 16B chunks per lane.
DEVFN v16bf load_bT16x32(const __bf16* base, int ld, int kb) {
  const int lane = threadIdx.x & 31;
  const int hf = lane >> 4;
  const int r  = lane & 15;
  const __bf16* p = base + (size_t)r * ld + kb + hf * 8;
  v8bf lo = *reinterpret_cast<const v8bf*>(p);
  v8bf hi = *reinterpret_cast<const v8bf*>(p + 16);
  v16bf out;
#pragma unroll
  for (int i = 0; i < 8; ++i) { out[i] = lo[i]; out[8 + i] = hi[i]; }
  return out;
}

DEVFN v8f wmma_bf16(v16bf a, v16bf b, v8f c) {
  return __builtin_amdgcn_wmma_f32_16x16x32_bf16(
      false, a, false, b, (short)0, c, false, false);
}

DEVFN unsigned lds_off(const void* p) {
  // Generic LDS pointer: low 32 bits carry the LDS byte offset.
  return (unsigned)(unsigned long long)(size_t)p;
}

// ---------------------------------------------------------------------------
// Tensor Data Mover: 2D tile load (global -> LDS), 2-byte elements.
// Builds D# group0 (count/lds/global addr/type=2) + group1 (dims/strides)
// per CDNA5 ISA and issues tensor_load_to_lds with
// VADDR2/3 = NULL (2D form).  Tracked by TENSORcnt.
// ---------------------------------------------------------------------------
DEVFN void tdm_load_2d_bf16(unsigned lds_addr, unsigned long long gaddr,
                            unsigned tensor_d0, unsigned tensor_d1,
                            unsigned tile_d0, unsigned tile_d1,
                            unsigned long long row_stride_elems) {
  v4u g0;
  g0[0] = 1u;                                      // count = 1 valid descriptor
  g0[1] = lds_addr;                                // lds_addr [63:32]
  g0[2] = (unsigned)(gaddr & 0xFFFFFFFFull);       // global_addr low
  g0[3] = (unsigned)((gaddr >> 32) & 0x01FFFFFFull) | (2u << 30);  // type=2
  v8u g1;
  g1[0] = (1u << 16);                              // data_size = 1 -> 2 bytes
  g1[1] = (tensor_d0 & 0xFFFFu) << 16;             // tensor_dim0[15:0] @bit48
  g1[2] = (tensor_d0 >> 16) | ((tensor_d1 & 0xFFFFu) << 16);
  g1[3] = (tensor_d1 >> 16) | (tile_d0 << 16);     // tile_dim0 @bit112
  g1[4] = tile_d1;                                 // tile_dim1; tile_dim2 = 0
  g1[5] = (unsigned)(row_stride_elems & 0xFFFFFFFFull);
  g1[6] = (unsigned)((row_stride_elems >> 32) & 0xFFFFull);  // stride1 = 0
  g1[7] = 0u;
  asm volatile("tensor_load_to_lds %0, %1" :: "s"(g0), "s"(g1) : "memory");
}

// ---------------------------------------------------------------------------
// fp32 -> bf16 conversion kernels
// ---------------------------------------------------------------------------
__global__ __launch_bounds__(256)
void cvt_bf16_kernel(const float* __restrict__ x, __bf16* __restrict__ y, int n) {
  int idx = blockIdx.x * 256 + threadIdx.x;
  if (idx < n) y[idx] = (__bf16)x[idx];
}

// W[K][N] fp32 -> Wt[N][K] bf16 (so B-fragments are K-contiguous)
__global__ __launch_bounds__(256)
void cvt_bf16_t_kernel(const float* __restrict__ w, __bf16* __restrict__ wt,
                       int K, int N) {
  int idx = blockIdx.x * 256 + threadIdx.x;
  if (idx >= K * N) return;
  int k = idx / N, n = idx % N;
  wt[(size_t)n * K + k] = (__bf16)w[idx];
}

// ---------------------------------------------------------------------------
// GEMM with TDM-staged A panel:
// C[M][N] = A[M][K] @ W[K][N] + bias  (A bf16 row-major, Bt bf16 [N][K]).
// Block: 256 thr = 8 waves (2 x 4), block tile 64x128, wave tile 32x32
// (2 A-frags x 2 B-frags -> 4 WMMAs per k-step, 16 FLOP/B register reuse).
// A panel (64 x 128 per chunk) double-buffered in LDS via tensor_load_to_lds;
// wave 0 drives the TDM and overlaps the next chunk with compute.
// ---------------------------------------------------------------------------
__global__ __launch_bounds__(256)
void gemm_bf16_bias_tdm_kernel(const __bf16* __restrict__ A,
                               const __bf16* __restrict__ Bt,
                               const float* __restrict__ bias,
                               float* __restrict__ C,
                               int M, int N, int K) {
  __shared__ __bf16 sA[2][64 * 128];               // 2 x 16KB double buffer

  const int wave = threadIdx.x >> 5;
  const int wm = wave & 1, wn = wave >> 1;
  const int mBlk = blockIdx.x * 64;
  const int n0 = blockIdx.y * 128 + wn * 32;
  const __bf16* brow0 = Bt + (size_t)n0 * K;
  const __bf16* brow1 = Bt + (size_t)(n0 + 16) * K;

  const int nChunks = K / 128;
  if (wave == 0) {
    tdm_load_2d_bf16(lds_off(&sA[0][0]),
                     (unsigned long long)(size_t)(A + (size_t)mBlk * K),
                     (unsigned)K, 64u, 128u, 64u, (unsigned long long)K);
  }

  v8f acc00 = {}, acc01 = {}, acc10 = {}, acc11 = {};

  for (int c = 0; c < nChunks; ++c) {
    const int cur = c & 1;
    if (wave == 0) {
      if (c + 1 < nChunks) {
        tdm_load_2d_bf16(lds_off(&sA[1 - cur][0]),
                         (unsigned long long)(size_t)(A + (size_t)mBlk * K +
                                                      (size_t)(c + 1) * 128),
                         (unsigned)K, 64u, 128u, 64u, (unsigned long long)K);
        __builtin_amdgcn_s_wait_tensorcnt(1);      // chunk c landed, c+1 in flight
      } else {
        __builtin_amdgcn_s_wait_tensorcnt(0);
      }
    }
    __syncthreads();                               // chunk c visible to all waves

    const __bf16* a0base = &sA[cur][(wm * 32) * 128];
    const __bf16* a1base = &sA[cur][(wm * 32 + 16) * 128];
    const int kg = c * 128;
    __builtin_prefetch(brow0 + (size_t)(threadIdx.x & 15) * K + kg + 128, 0, 0);
#pragma unroll
    for (int kk = 0; kk < 128; kk += 32) {
      v16bf a0 = load_a16x32(a0base, 128, kk);
      v16bf a1 = load_a16x32(a1base, 128, kk);
      v16bf b0 = load_bT16x32(brow0, K, kg + kk);
      v16bf b1 = load_bT16x32(brow1, K, kg + kk);
      acc00 = wmma_bf16(a0, b0, acc00);
      acc01 = wmma_bf16(a0, b1, acc01);
      acc10 = wmma_bf16(a1, b0, acc10);
      acc11 = wmma_bf16(a1, b1, acc11);
    }
    __syncthreads();                               // done reading; TDM may reuse
  }

  const int lane = threadIdx.x & 31;
  const int hf = lane >> 4, col = lane & 15;
  const int m0 = mBlk + wm * 32;
  const float b0s = bias[n0 + col];
  const float b1s = bias[n0 + 16 + col];
#pragma unroll
  for (int i = 0; i < 8; ++i) {
    int r0 = m0 + i + 8 * hf;
    C[(size_t)r0 * N + n0 + col]              = acc00[i] + b0s;
    C[(size_t)r0 * N + n0 + 16 + col]         = acc01[i] + b1s;
    C[(size_t)(r0 + 16) * N + n0 + col]       = acc10[i] + b0s;
    C[(size_t)(r0 + 16) * N + n0 + 16 + col]  = acc11[i] + b1s;
  }
}

// ---------------------------------------------------------------------------
// RMSNorm (+ optional RoPE) and head-major bf16 packing: dst[H][Trows][DH]
// ---------------------------------------------------------------------------
__global__ __launch_bounds__(256)
void rms_rope_pack_kernel(const float* __restrict__ x, const float* __restrict__ g,
                          const float* __restrict__ rcos, const float* __restrict__ rsin,
                          __bf16* __restrict__ dst, int Trows, int do_rope) {
  const int t = blockIdx.x;
  const float* row = x + (size_t)t * kDIM;

  float ss = 0.f;
  for (int c = threadIdx.x; c < kDIM; c += 256) { float v = row[c]; ss += v * v; }
#pragma unroll
  for (int m = 16; m >= 1; m >>= 1) ss += __shfl_xor(ss, m, 32);

  __shared__ float part[8];
  __shared__ float sscale;
  if ((threadIdx.x & 31) == 0) part[threadIdx.x >> 5] = ss;
  __syncthreads();
  if (threadIdx.x == 0) {
    float s = 0.f;
#pragma unroll
    for (int i = 0; i < 8; ++i) s += part[i];
    sscale = rsqrtf(s / (float)kDIM + 1e-6f);
  }
  __syncthreads();
  const float scale = sscale;

  if (do_rope) {
    for (int p = threadIdx.x; p < kDIM / 2; p += 256) {
      int h = p >> 6, pp = p & 63;
      int ce = h * kDH + 2 * pp;
      float xe = row[ce]     * scale * g[ce];
      float xo = row[ce + 1] * scale * g[ce + 1];
      float c = rcos[(size_t)t * (kDH / 2) + pp];
      float s = rsin[(size_t)t * (kDH / 2) + pp];
      size_t o = ((size_t)h * Trows + t) * kDH + 2 * pp;
      dst[o]     = (__bf16)(xe * c - xo * s);
      dst[o + 1] = (__bf16)(xe * s + xo * c);
    }
  } else {
    for (int c = threadIdx.x; c < kDIM; c += 256) {
      int h = c >> 7, d = c & 127;
      dst[((size_t)h * Trows + t) * kDH + d] = (__bf16)(row[c] * scale * g[c]);
    }
  }
}

// V pack transposed: v_lin[T][DIM] fp32 -> Vt[H][DH][Trows] bf16
__global__ __launch_bounds__(256)
void pack_v_kernel(const float* __restrict__ v, __bf16* __restrict__ vt, int Trows) {
  int idx = blockIdx.x * 256 + threadIdx.x;
  if (idx >= Trows * kDIM) return;
  int t = idx / kDIM, c = idx % kDIM;       // c == h*DH + d
  vt[(size_t)c * Trows + t] = (__bf16)v[idx];
}

// ---------------------------------------------------------------------------
// Flash attention (bf16 WMMA, fp32 online softmax).
// Q[H][Tq][DH], K[H][Tk][DH], Vt[H][DH][Tk]; Out fp32 [Tq][DIM].
// 128 threads = 4 waves; each wave: 16 query rows; key tiles of 32.
// ---------------------------------------------------------------------------
__global__ __launch_bounds__(128)
void flash_attn_kernel(const __bf16* __restrict__ Q, const __bf16* __restrict__ Km,
                       const __bf16* __restrict__ Vt, float* __restrict__ Out,
                       int Tq, int Tk, float scale, int accumulate) {
  __shared__ __bf16 sm[4][16 * 32];

  const int wave = threadIdx.x >> 5;
  const int lane = threadIdx.x & 31;
  const int hf = lane >> 4, ln = lane & 15;
  const int head = blockIdx.y;
  const int q0 = blockIdx.x * 64 + wave * 16;

  const __bf16* qbase = Q  + ((size_t)head * Tq + q0) * kDH;
  const __bf16* kbase = Km + (size_t)head * Tk * kDH;
  const __bf16* vbase = Vt + (size_t)head * kDH * (size_t)Tk;
  __bf16* myLds = &sm[wave][0];

  v16bf aq[4];
#pragma unroll
  for (int kk = 0; kk < 4; ++kk) aq[kk] = load_a16x32(qbase, kDH, kk * 32);

  v8f o[8];
#pragma unroll
  for (int f = 0; f < 8; ++f) o[f] = v8f{};
  float m_[8], l_[8];
#pragma unroll
  for (int i = 0; i < 8; ++i) { m_[i] = -1e30f; l_[i] = 0.f; }

  for (int kb = 0; kb < Tk; kb += 32) {
    v8f s0 = {}, s1 = {};
    const __bf16* k0 = kbase + (size_t)kb * kDH;
    const __bf16* k1 = kbase + (size_t)(kb + 16) * kDH;
#pragma unroll
    for (int kk = 0; kk < 4; ++kk) {
      s0 = wmma_bf16(aq[kk], load_bT16x32(k0, kDH, kk * 32), s0);
      s1 = wmma_bf16(aq[kk], load_bT16x32(k1, kDH, kk * 32), s1);
    }

    float mt[8];
#pragma unroll
    for (int i = 0; i < 8; ++i) {
      s0[i] *= scale; s1[i] *= scale;
      mt[i] = fmaxf(s0[i], s1[i]);
    }
#pragma unroll
    for (int msk = 8; msk >= 1; msk >>= 1) {
#pragma unroll
      for (int i = 0; i < 8; ++i) mt[i] = fmaxf(mt[i], __shfl_xor(mt[i], msk, 32));
    }
    float cf[8], rs[8];
#pragma unroll
    for (int i = 0; i < 8; ++i) {
      float mn = fmaxf(m_[i], mt[i]);
      cf[i] = __expf(m_[i] - mn);
      m_[i] = mn;
      float p0 = __expf(s0[i] - mn);
      float p1 = __expf(s1[i] - mn);
      s0[i] = p0; s1[i] = p1;
      rs[i] = p0 + p1;
    }
#pragma unroll
    for (int msk = 8; msk >= 1; msk >>= 1) {
#pragma unroll
      for (int i = 0; i < 8; ++i) rs[i] += __shfl_xor(rs[i], msk, 32);
    }
#pragma unroll
    for (int i = 0; i < 8; ++i) l_[i] = l_[i] * cf[i] + rs[i];
#pragma unroll
    for (int f = 0; f < 8; ++f) {
#pragma unroll
      for (int i = 0; i < 8; ++i) o[f][i] *= cf[i];
    }

    __syncthreads();
#pragma unroll
    for (int i = 0; i < 8; ++i) {
      int row = i + 8 * hf;
      myLds[row * 32 + ln]      = (__bf16)s0[i];
      myLds[row * 32 + 16 + ln] = (__bf16)s1[i];
    }
    __syncthreads();
    v16bf ap = load_a16x32(myLds, 32, 0);

#pragma unroll
    for (int f = 0; f < 8; ++f)
      o[f] = wmma_bf16(ap, load_bT16x32(vbase + (size_t)(f * 16) * Tk, Tk, kb), o[f]);
  }

#pragma unroll
  for (int f = 0; f < 8; ++f) {
#pragma unroll
    for (int i = 0; i < 8; ++i) {
      size_t oidx = (size_t)(q0 + i + 8 * hf) * kDIM + (size_t)head * kDH + f * 16 + ln;
      float val = o[f][i] / l_[i];
      Out[oidx] = accumulate ? (Out[oidx] + val) : val;
    }
  }
}

// ---------------------------------------------------------------------------
// Host orchestration
// ---------------------------------------------------------------------------
extern "C" void kernel_launch(void* const* d_in, const int* in_sizes, int n_in,
                              void* d_out, int out_size, void* d_ws, size_t ws_size,
                              hipStream_t stream) {
  (void)in_sizes; (void)n_in; (void)out_size; (void)ws_size;
  const float* hs   = (const float*)d_in[0];
  const float* rhs  = (const float*)d_in[1];
  const float* rcos = (const float*)d_in[2];
  const float* rsin = (const float*)d_in[3];
  const float* Wq  = (const float*)d_in[4];  const float* bq  = (const float*)d_in[5];
  const float* Wk  = (const float*)d_in[6];  const float* bk  = (const float*)d_in[7];
  const float* Wv  = (const float*)d_in[8];  const float* bv  = (const float*)d_in[9];
  const float* Wkr = (const float*)d_in[10]; const float* bkr = (const float*)d_in[11];
  const float* Wvr = (const float*)d_in[12]; const float* bvr = (const float*)d_in[13];
  const float* Wo  = (const float*)d_in[14]; const float* bo  = (const float*)d_in[15];
  const float* gq  = (const float*)d_in[16]; const float* gk  = (const float*)d_in[17];
  float* out = (float*)d_out;

  char* ws = (char*)d_ws;
  size_t off = 0;
  auto take = [&](size_t bytes) -> char* {
    char* p = ws + off;
    off = (off + bytes + 255) & ~(size_t)255;
    return p;
  };

  const size_t szWt  = (size_t)kDIM * kDIM * sizeof(__bf16);
  __bf16* wtq  = (__bf16*)take(szWt);
  __bf16* wtk  = (__bf16*)take(szWt);
  __bf16* wtv  = (__bf16*)take(szWt);
  __bf16* wtkr = (__bf16*)take(szWt);
  __bf16* wtvr = (__bf16*)take(szWt);
  __bf16* wto  = (__bf16*)take(szWt);

  __bf16* hsb  = (__bf16*)take((size_t)kT  * kDIM * sizeof(__bf16));
  __bf16* refb = (__bf16*)take((size_t)kTR * kDIM * sizeof(__bf16));

  float* qlin  = (float*)take((size_t)kT  * kDIM * sizeof(float));
  float* klin  = (float*)take((size_t)kT  * kDIM * sizeof(float));
  float* vlin  = (float*)take((size_t)kT  * kDIM * sizeof(float));
  float* krlin = (float*)take((size_t)kTR * kDIM * sizeof(float));
  float* vrlin = (float*)take((size_t)kTR * kDIM * sizeof(float));

  __bf16* Qm  = (__bf16*)take((size_t)kT  * kDIM * sizeof(__bf16));
  __bf16* Km  = (__bf16*)take((size_t)kT  * kDIM * sizeof(__bf16));
  __bf16* Qr  = (__bf16*)take((size_t)kT  * kDIM * sizeof(__bf16));
  __bf16* Kr  = (__bf16*)take((size_t)kTR * kDIM * sizeof(__bf16));
  __bf16* Vtm = (__bf16*)take((size_t)kT  * kDIM * sizeof(__bf16));
  __bf16* Vtr = (__bf16*)take((size_t)kTR * kDIM * sizeof(__bf16));

  float*  asum = (float*)take((size_t)kT * kDIM * sizeof(float));
  __bf16* abf  = (__bf16*)take((size_t)kT * kDIM * sizeof(__bf16));

  const int nHS  = kT * kDIM;
  const int nREF = kTR * kDIM;
  const int nW   = kDIM * kDIM;

  // 1) conversions
  cvt_bf16_kernel<<<(nHS + 255) / 256, 256, 0, stream>>>(hs, hsb, nHS);
  cvt_bf16_kernel<<<(nREF + 255) / 256, 256, 0, stream>>>(rhs, refb, nREF);
  cvt_bf16_t_kernel<<<(nW + 255) / 256, 256, 0, stream>>>(Wq,  wtq,  kDIM, kDIM);
  cvt_bf16_t_kernel<<<(nW + 255) / 256, 256, 0, stream>>>(Wk,  wtk,  kDIM, kDIM);
  cvt_bf16_t_kernel<<<(nW + 255) / 256, 256, 0, stream>>>(Wv,  wtv,  kDIM, kDIM);
  cvt_bf16_t_kernel<<<(nW + 255) / 256, 256, 0, stream>>>(Wkr, wtkr, kDIM, kDIM);
  cvt_bf16_t_kernel<<<(nW + 255) / 256, 256, 0, stream>>>(Wvr, wtvr, kDIM, kDIM);
  cvt_bf16_t_kernel<<<(nW + 255) / 256, 256, 0, stream>>>(Wo,  wto,  kDIM, kDIM);

  // 2) projection GEMMs (WMMA bf16, TDM-staged A panel)
  dim3 gemm_blk(256);
  dim3 g_main(kT / 64, kDIM / 128);
  dim3 g_ref(kTR / 64, kDIM / 128);
  gemm_bf16_bias_tdm_kernel<<<g_main, gemm_blk, 0, stream>>>(hsb,  wtq,  bq,  qlin,  kT,  kDIM, kDIM);
  gemm_bf16_bias_tdm_kernel<<<g_main, gemm_blk, 0, stream>>>(hsb,  wtk,  bk,  klin,  kT,  kDIM, kDIM);
  gemm_bf16_bias_tdm_kernel<<<g_main, gemm_blk, 0, stream>>>(hsb,  wtv,  bv,  vlin,  kT,  kDIM, kDIM);
  gemm_bf16_bias_tdm_kernel<<<g_ref,  gemm_blk, 0, stream>>>(refb, wtkr, bkr, krlin, kTR, kDIM, kDIM);
  gemm_bf16_bias_tdm_kernel<<<g_ref,  gemm_blk, 0, stream>>>(refb, wtvr, bvr, vrlin, kTR, kDIM, kDIM);

  // 3) rmsnorm (+rope) packing and V transpose packing
  rms_rope_pack_kernel<<<kT,  256, 0, stream>>>(qlin,  gq, rcos, rsin, Qm, kT,  1);
  rms_rope_pack_kernel<<<kT,  256, 0, stream>>>(klin,  gk, rcos, rsin, Km, kT,  1);
  rms_rope_pack_kernel<<<kT,  256, 0, stream>>>(qlin,  gq, rcos, rsin, Qr, kT,  0);
  rms_rope_pack_kernel<<<kTR, 256, 0, stream>>>(krlin, gk, rcos, rsin, Kr, kTR, 0);
  pack_v_kernel<<<(nHS + 255) / 256, 256, 0, stream>>>(vlin,  Vtm, kT);
  pack_v_kernel<<<(nREF + 255) / 256, 256, 0, stream>>>(vrlin, Vtr, kTR);

  // 4) attentions: main writes, ref accumulates (REF_SCALE = 1.0)
  const float scale = 0.08838834764831845f;  // 1/sqrt(128)
  dim3 fa_grid(kT / 64, kH);
  flash_attn_kernel<<<fa_grid, 128, 0, stream>>>(Qm, Km, Vtm, asum, kT, kT,  scale, 0);
  flash_attn_kernel<<<fa_grid, 128, 0, stream>>>(Qr, Kr, Vtr, asum, kT, kTR, scale, 1);

  // 5) output projection
  cvt_bf16_kernel<<<(nHS + 255) / 256, 256, 0, stream>>>(asum, abf, nHS);
  gemm_bf16_bias_tdm_kernel<<<g_main, gemm_blk, 0, stream>>>(abf, wto, bo, out, kT, kDIM, kDIM);
}
